// GNNModel_74801150427817
// MI455X (gfx1250) — matrix operand order repaired and checked by reading, not compiled, not run on hardware
//
#include <hip/hip_runtime.h>
#include <hip/hip_bf16.h>

typedef __attribute__((ext_vector_type(16))) __bf16 v16bf;
typedef __attribute__((ext_vector_type(8)))  __bf16 v8bf;
typedef __attribute__((ext_vector_type(8)))  float  v8f;

#define HEADS 4
#define OUTC 32
#define CF 128            // HEADS*OUTC == C_IN == 128
#define NEG_SLOPE 0.2f
#define MT 4              // row sub-tiles (16 rows each) per block -> 64 rows/block

union BU { v16bf v; v8bf p[2]; };

// ---------- helpers ----------
__device__ __forceinline__ float lrelu(float e) { return e < 0.f ? e * NEG_SLOPE : e; }
// monotone encoding of float for unsigned atomicMax (total order, handles negatives)
__device__ __forceinline__ unsigned enc_f(float f) {
    unsigned u = __float_as_uint(f);
    return (u >> 31) ? ~u : (u | 0x80000000u);
}
__device__ __forceinline__ float dec_f(unsigned u) {
    return __uint_as_float((u >> 31) ? (u & 0x7fffffffu) : ~u);
}

// ---------- 0) fp32 -> bf16 conversion; W is stored transposed (wt[n][k]) ----------
__global__ void convert_k(const float* __restrict__ x, const float* __restrict__ W,
                          __bf16* __restrict__ xb, __bf16* __restrict__ wt, long nx) {
    long t = (long)blockIdx.x * blockDim.x + threadIdx.x;
    if (t < nx) xb[t] = (__bf16)x[t];
    if (t < CF * CF) {
        int k = (int)(t >> 7), n = (int)(t & 127);
        wt[(long)n * CF + k] = (__bf16)W[t];
    }
}

// ---------- 1) h = x @ W via v_wmma_f32_16x16x32_bf16 ----------
// block = 256 threads = 8 waves; block covers 64 rows x 128 cols.
// wave w -> cols [16w,16w+16); B fragments (4 k-steps) preloaded once, reused for MT row tiles.
__global__ void gemm_bf16_k(const __bf16* __restrict__ xb, const __bf16* __restrict__ wt,
                            float* __restrict__ hmat, int nrows) {
    const int wave = threadIdx.x >> 5;
    const int lane = threadIdx.x & 31;
    const int hi   = lane >> 4;      // half-wave select
    const int l    = lane & 15;
    const int col  = wave * 16 + l;  // B column / C column owned by this lane

    // B 32x16 bf16 (K-major per column): lanes 0-15 hold K=kk+[0..7] & kk+[16..23]; lanes 16-31 +8..15 & +24..31
    const __bf16* bcol = wt + (long)col * CF;
    BU B[4];
    #pragma unroll
    for (int ks = 0; ks < 4; ++ks) {
        const int boff = ks * 32 + hi * 8;
        B[ks].p[0] = *(const v8bf*)(bcol + boff);
        B[ks].p[1] = *(const v8bf*)(bcol + boff + 16);
    }

    const long rowbase = (long)blockIdx.x * (16 * MT);
    v8f c[MT] = {};

    #pragma unroll
    for (int mt = 0; mt < MT; ++mt) {
        long row = rowbase + mt * 16 + l;          // A row owned by this lane
        if (row >= nrows) row = nrows - 1;         // clamp (dup rows OK for loads)
        const __bf16* arow = xb + row * CF;
        #pragma unroll
        for (int ks = 0; ks < 4; ++ks) {
            // A 16x32 bf16: lanes 0-15 hold K=kk+[0..7] & kk+[16..23]; lanes 16-31 +8..15 & +24..31
            BU A;
            const int aoff = ks * 32 + hi * 8;
            A.p[0] = *(const v8bf*)(arow + aoff);
            A.p[1] = *(const v8bf*)(arow + aoff + 16);
            c[mt] = __builtin_amdgcn_wmma_f32_16x16x32_bf16(false, A.v, false, B[ks].v,
                                                            (short)0, c[mt], false, false);
        }
    }

    // C/D layout: VGPR r -> M = r + 8*hi ; N = lane&15.
    // Fast path: whole 8-row strip in range -> unconditional stores.
    #pragma unroll
    for (int mt = 0; mt < MT; ++mt) {
        const long mb = rowbase + mt * 16 + hi * 8;
        if (mb + 7 < nrows) {
            float* dst = hmat + mb * CF + col;
            #pragma unroll
            for (int r = 0; r < 8; ++r) dst[(long)r * CF] = c[mt][r];
        } else {
            #pragma unroll
            for (int r = 0; r < 8; ++r) {
                long m = mb + r;
                if (m < nrows) hmat[m * CF + col] = c[mt][r];
            }
        }
    }
}

// ---------- 2) per-node attention scores ----------
__global__ void scores_k(const float* __restrict__ hmat, const float* __restrict__ att_src,
                         const float* __restrict__ att_dst, float* __restrict__ as_,
                         float* __restrict__ ad_, int N) {
    int n = blockIdx.x * blockDim.x + threadIdx.x;
    if (n >= N) return;
    const float4* hr = (const float4*)(hmat + (long)n * CF);
    #pragma unroll
    for (int hh = 0; hh < HEADS; ++hh) {
        const float4* s4 = (const float4*)(att_src + hh * OUTC);
        const float4* d4 = (const float4*)(att_dst + hh * OUTC);
        float s = 0.f, d = 0.f;
        #pragma unroll
        for (int q = 0; q < OUTC / 4; ++q) {
            float4 v = hr[hh * (OUTC / 4) + q];
            float4 a = s4[q], b = d4[q];
            s += v.x * a.x + v.y * a.y + v.z * a.z + v.w * a.w;
            d += v.x * b.x + v.y * b.y + v.z * b.z + v.w * b.w;
        }
        as_[n * HEADS + hh] = s;
        ad_[n * HEADS + hh] = d;
    }
}

// ---------- 3) init accumulators (graph replay safe) ----------
__global__ void init_k(unsigned* __restrict__ emax, float* __restrict__ esum,
                       float* __restrict__ accum, int N) {
    long t = (long)blockIdx.x * blockDim.x + threadIdx.x;
    long n4 = (long)N * HEADS;
    if (t < n4) { emax[t] = 0u; esum[t] = 0.f; }
    if (t < (long)N * CF) accum[t] = 0.f;
}

// ---------- 4) segment max over incoming edges ----------
__global__ void edge_max_k(const int* __restrict__ ei, int E, int N,
                           const float* __restrict__ as_, const float* __restrict__ ad_,
                           unsigned* __restrict__ emax) {
    int t = blockIdx.x * blockDim.x + threadIdx.x;
    if (t >= E + N) return;
    int s, d;
    if (t < E) { s = ei[t]; d = ei[E + t]; } else { s = t - E; d = s; }   // self loops
    #pragma unroll
    for (int hh = 0; hh < HEADS; ++hh) {
        float e = lrelu(as_[s * HEADS + hh] + ad_[d * HEADS + hh]);
        atomicMax(&emax[d * HEADS + hh], enc_f(e));
    }
}

// ---------- 5) segment sum of exp ----------
__global__ void edge_sum_k(const int* __restrict__ ei, int E, int N,
                           const float* __restrict__ as_, const float* __restrict__ ad_,
                           const unsigned* __restrict__ emax, float* __restrict__ esum) {
    int t = blockIdx.x * blockDim.x + threadIdx.x;
    if (t >= E + N) return;
    int s, d;
    if (t < E) { s = ei[t]; d = ei[E + t]; } else { s = t - E; d = s; }
    #pragma unroll
    for (int hh = 0; hh < HEADS; ++hh) {
        float e = lrelu(as_[s * HEADS + hh] + ad_[d * HEADS + hh]);
        float m = dec_f(emax[d * HEADS + hh]);
        atomicAdd(&esum[d * HEADS + hh], __expf(e - m));
    }
}

// ---------- 6) weighted aggregation: accum[dst] += alpha * h[src] ----------
__global__ void edge_agg_k(const int* __restrict__ ei, int E, int N,
                           const float* __restrict__ as_, const float* __restrict__ ad_,
                           const unsigned* __restrict__ emax, const float* __restrict__ esum,
                           const float* __restrict__ hmat, float* __restrict__ accum) {
    long t = (long)blockIdx.x * blockDim.x + threadIdx.x;
    long tot = (long)(E + N) * CF;
    if (t >= tot) return;
    int  f  = (int)(t & (CF - 1));
    long e  = t >> 7;
    int  hh = f >> 5;
    int s, d;
    if (e < E) { s = ei[e]; d = ei[E + e]; } else { s = (int)(e - E); d = s; }
    float sc = lrelu(as_[s * HEADS + hh] + ad_[d * HEADS + hh]);
    float m  = dec_f(emax[d * HEADS + hh]);
    float alpha = __expf(sc - m) / esum[d * HEADS + hh];
    atomicAdd(&accum[(long)d * CF + f], hmat[(long)s * CF + f] * alpha);
}

// ---------- 7) bias + relu + fc + sigmoid ----------
__global__ void final_k(const float* __restrict__ accum, const float* __restrict__ bias,
                        const float* __restrict__ fcw, const float* __restrict__ fcb,
                        float* __restrict__ out, int N) {
    int n = blockIdx.x * blockDim.x + threadIdx.x;
    if (n >= N) return;
    const float4* a4 = (const float4*)(accum + (long)n * CF);
    const float4* b4 = (const float4*)bias;
    const float4* w4 = (const float4*)fcw;
    float acc = fcb[0];
    #pragma unroll
    for (int q = 0; q < CF / 4; ++q) {
        float4 v = a4[q], b = b4[q], w = w4[q];
        acc += fmaxf(v.x + b.x, 0.f) * w.x;
        acc += fmaxf(v.y + b.y, 0.f) * w.y;
        acc += fmaxf(v.z + b.z, 0.f) * w.z;
        acc += fmaxf(v.w + b.w, 0.f) * w.w;
    }
    out[n] = 1.f / (1.f + __expf(-acc));
}

extern "C" void kernel_launch(void* const* d_in, const int* in_sizes, int n_in,
                              void* d_out, int out_size, void* d_ws, size_t ws_size,
                              hipStream_t stream) {
    const float* x       = (const float*)d_in[0];
    const int*   ei      = (const int*)d_in[1];
    const float* W       = (const float*)d_in[2];
    const float* att_src = (const float*)d_in[3];
    const float* att_dst = (const float*)d_in[4];
    const float* bias    = (const float*)d_in[5];
    const float* fc_w    = (const float*)d_in[6];
    const float* fc_b    = (const float*)d_in[7];
    float* out = (float*)d_out;

    const int N = in_sizes[0] / CF;
    const int E = in_sizes[1] / 2;

    // workspace carve-up (256B aligned)
    char* base = (char*)d_ws;
    size_t off = 0;
    auto carve = [&](size_t bytes) { char* p = base + off; off = (off + bytes + 255) & ~(size_t)255; return p; };
    __bf16*   xb    = (__bf16*)  carve((size_t)N * CF * sizeof(__bf16));
    __bf16*   wt    = (__bf16*)  carve((size_t)CF * CF * sizeof(__bf16));
    float*    hmat  = (float*)   carve((size_t)N * CF * sizeof(float));
    float*    as_   = (float*)   carve((size_t)N * HEADS * sizeof(float));
    float*    ad_   = (float*)   carve((size_t)N * HEADS * sizeof(float));
    unsigned* emax  = (unsigned*)carve((size_t)N * HEADS * sizeof(unsigned));
    float*    esum  = (float*)   carve((size_t)N * HEADS * sizeof(float));
    float*    accum = (float*)   carve((size_t)N * CF * sizeof(float));
    (void)ws_size;

    const int BS = 256;
    long nx = (long)N * CF;

    convert_k<<<(unsigned)((nx + BS - 1) / BS), BS, 0, stream>>>(x, W, xb, wt, nx);

    gemm_bf16_k<<<(unsigned)((N + 16 * MT - 1) / (16 * MT)), BS, 0, stream>>>(xb, wt, hmat, N);

    scores_k<<<(unsigned)((N + BS - 1) / BS), BS, 0, stream>>>(hmat, att_src, att_dst, as_, ad_, N);

    init_k<<<(unsigned)((nx + BS - 1) / BS), BS, 0, stream>>>(emax, esum, accum, N);

    const int Et = E + N;
    edge_max_k<<<(unsigned)((Et + BS - 1) / BS), BS, 0, stream>>>(ei, E, N, as_, ad_, emax);
    edge_sum_k<<<(unsigned)((Et + BS - 1) / BS), BS, 0, stream>>>(ei, E, N, as_, ad_, emax, esum);

    long tot = (long)Et * CF;
    edge_agg_k<<<(unsigned)((tot + BS - 1) / BS), BS, 0, stream>>>(ei, E, N, as_, ad_, emax, esum, hmat, accum);

    final_k<<<(unsigned)((N + BS - 1) / BS), BS, 0, stream>>>(accum, bias, fc_w, fc_b, out, N);
}
